// MambaTCNRegNet_25417616457864
// MI455X (gfx1250) — matrix operand order repaired and checked by reading, not compiled, not run on hardware
//
#include <hip/hip_runtime.h>
#include <cstdint>
#include <cstddef>

// ---------------------------------------------------------------------------
// MambaTCNRegNet forward for gfx1250 (MI455X), wave32 + WMMA bf16.
// - All GEMM-shaped math -> v_wmma_f32_16x16x32_bf16 (f32 accumulate).
// - Fast GEMM path: guard-free, B tile staged to LDS via async global->LDS
//   (ASYNCcnt), 32x64 per wave (8 WMMAs / k-step).
// - Guarded GEMM path: ragged N (260) and shifted causal conv taps.
// - Dilated causal convs = 5 shifted accumulating GEMMs (no im2col).
// - Attention = flash-style per (batch,head) with WMMA QK^T and PV.
// Param flattening assumption: setup_inputs() dict insertion order, recursive.
// ---------------------------------------------------------------------------

#define S_LEN 1024
#define BATCH 8
#define RROWS 8192   // S*B rows

typedef __attribute__((ext_vector_type(16))) __bf16   v16bf;
typedef __attribute__((ext_vector_type(8)))  float    v8f;
typedef __attribute__((ext_vector_type(4)))  unsigned u32x4;

struct Frag { u32x4 lo, hi; };

static __device__ __forceinline__ v16bf fragv(Frag f){ return __builtin_bit_cast(v16bf, f); }

static __device__ __forceinline__ unsigned short f2bfu(float f){
  unsigned u = __builtin_bit_cast(unsigned, f);
  unsigned r = u + 0x7FFFu + ((u >> 16) & 1u);   // round-to-nearest-even
  return (unsigned short)(r >> 16);
}
static __device__ __forceinline__ float bf2f(unsigned short h){
  unsigned u = ((unsigned)h) << 16;
  return __builtin_bit_cast(float, u);
}
static __device__ __forceinline__ float geluf(float x){
  return 0.5f * x * (1.f + erff(x * 0.7071067811865475f));
}
static __device__ __forceinline__ float sigm(float x){ return 1.f / (1.f + __expf(-x)); }

// ---------------------------------------------------------------------------
// FAST GEMM: requires M%256==0, N%64==0, rowShift==0, lda%8==0, K%32==0.
// Block = 8 waves; block tile 256(M) x 64(N); wave tile 32x64.
// B tile (64n x 32k bf16 = 4KB) staged in LDS with async global->LDS loads.
// ---------------------------------------------------------------------------
__global__ void __launch_bounds__(256) k_gemm_fast(
    const unsigned short* __restrict__ A, int lda,
    const unsigned short* __restrict__ WT,
    const float* __restrict__ bias,
    float* __restrict__ C, int ldc,
    int K, int accum)
{
  __shared__ unsigned short Bsh[64 * 32];      // 4KB
  const int tid  = threadIdx.x;
  const int lane = tid & 31;
  const int w    = tid >> 5;
  const int half = lane >> 4;
  const int r16  = lane & 15;
  const int m0   = blockIdx.y * 256 + w * 32;
  const int n0   = blockIdx.x * 64;

  v8f acc[8];
  #pragma unroll
  for (int j = 0; j < 8; j++) acc[j] = v8f{};

  const unsigned short* a0p = A + (size_t)(m0 + r16) * (size_t)lda + half * 8;
  const unsigned short* a1p = a0p + (size_t)16 * (size_t)lda;

  // async B staging: each thread moves one 16B chunk of the 64x32 tile
  const int nloc = tid >> 2, part = tid & 3;
  const unsigned short* bsrc = WT + (size_t)(n0 + nloc) * (size_t)K + part * 8;
  unsigned short* bdst = &Bsh[nloc * 32 + part * 8];
  const unsigned bdst_lds = (unsigned)(size_t)bdst;   // low 32 bits = LDS offset

  for (int k0 = 0; k0 < K; k0 += 32){
    __syncthreads();                                   // tile consumed
    {
      unsigned long long ga = (unsigned long long)(bsrc + k0);
      asm volatile("global_load_async_to_lds_b128 %0, %1, off"
                   :: "v"(bdst_lds), "v"(ga) : "memory");
      asm volatile("s_wait_asynccnt 0" ::: "memory");
    }
    __syncthreads();                                   // tile ready (all waves)

    Frag fa0, fa1;
    { const unsigned short* p = a0p + k0;
      fa0.lo = *(const u32x4*)p; fa0.hi = *(const u32x4*)(p + 16); }
    { const unsigned short* p = a1p + k0;
      fa1.lo = *(const u32x4*)p; fa1.hi = *(const u32x4*)(p + 16); }
    if (k0 + 32 < K){
      __builtin_prefetch(a0p + k0 + 32, 0, 3);
      __builtin_prefetch(a1p + k0 + 32, 0, 3);
    }
    v16bf va0 = fragv(fa0), va1 = fragv(fa1);
    #pragma unroll
    for (int j = 0; j < 4; j++){
      Frag fb;
      const unsigned short* p = &Bsh[(j * 16 + r16) * 32 + half * 8];
      fb.lo = *(const u32x4*)p; fb.hi = *(const u32x4*)(p + 16);
      v16bf vb = fragv(fb);
      acc[j]     = __builtin_amdgcn_wmma_f32_16x16x32_bf16(false, va0, false, vb,
                                                           (short)0, acc[j],     false, false);
      acc[4 + j] = __builtin_amdgcn_wmma_f32_16x16x32_bf16(false, va1, false, vb,
                                                           (short)0, acc[4 + j], false, false);
    }
  }
  #pragma unroll
  for (int j = 0; j < 4; j++){
    const int col = n0 + j * 16 + r16;
    const float bv = bias ? bias[col] : 0.f;
    #pragma unroll
    for (int i = 0; i < 8; i++){
      float* c0 = C + (size_t)(m0 + i + half * 8) * (size_t)ldc + col;
      float* c1 = C + (size_t)(m0 + 16 + i + half * 8) * (size_t)ldc + col;
      if (accum){ *c0 += acc[j][i]; *c1 += acc[4 + j][i]; }
      else      { *c0  = acc[j][i] + bv; *c1  = acc[4 + j][i] + bv; }
    }
  }
}

// ---------------------------------------------------------------------------
// GUARDED GEMM: handles ragged N and causal row shifting (dilated conv taps):
//   source row = m - rowShift, valid iff (m % seg) >= rowShift (zero otherwise)
// ---------------------------------------------------------------------------
__global__ void __launch_bounds__(256) k_gemm_g(
    const unsigned short* __restrict__ A, int lda,
    const unsigned short* __restrict__ WT,
    const float* __restrict__ bias,
    float* __restrict__ C, int ldc,
    int M, int N, int K, int rowShift, int seg, int accum)
{
  const int lane = threadIdx.x & 31;
  const int w    = threadIdx.x >> 5;
  const int half = lane >> 4;
  const int r16  = lane & 15;
  const int m0   = blockIdx.y * 128 + w * 16;
  const int n0   = blockIdx.x * 64;

  v8f acc0{}, acc1{}, acc2{}, acc3{};

  const int  arow   = m0 + r16;
  const bool avalid = (arow < M) && ((arow % seg) >= rowShift);
  const unsigned short* arp =
      avalid ? (A + (size_t)(arow - rowShift) * (size_t)lda + half * 8) : (const unsigned short*)0;

  for (int k0 = 0; k0 < K; k0 += 32){
    Frag fa{};
    if (avalid){
      const unsigned short* p = arp + k0;
      fa.lo = *(const u32x4*)p;
      fa.hi = *(const u32x4*)(p + 16);
      if (k0 + 32 < K) __builtin_prefetch(p + 32, 0, 3);
    }
    v16bf va = fragv(fa);
    #pragma unroll
    for (int j = 0; j < 4; j++){
      const int n = n0 + j * 16 + r16;
      Frag fb{};
      if (n < N){
        const unsigned short* p = WT + (size_t)n * (size_t)K + k0 + half * 8;
        fb.lo = *(const u32x4*)p;
        fb.hi = *(const u32x4*)(p + 16);
      }
      v8f& acc = (j==0)?acc0:(j==1)?acc1:(j==2)?acc2:acc3;
      acc = __builtin_amdgcn_wmma_f32_16x16x32_bf16(false, va, false, fragv(fb),
                                                    (short)0, acc, false, false);
    }
  }
  #pragma unroll
  for (int j = 0; j < 4; j++){
    const int col = n0 + j * 16 + r16;
    if (col >= N) continue;
    const v8f acc = (j==0)?acc0:(j==1)?acc1:(j==2)?acc2:acc3;
    const float bv = bias ? bias[col] : 0.f;
    #pragma unroll
    for (int i = 0; i < 8; i++){
      const int row = m0 + i + half * 8;
      if (row < M){
        float* cp = C + (size_t)row * (size_t)ldc + col;
        if (accum) *cp += acc[i];
        else       *cp  = acc[i] + bv;
      }
    }
  }
}

// ---------------------------------------------------------------------------
// Flash attention per (b,h). Q:[s][32] (pre-scaled), K:[s][32], VT:[32][s] bf16
// AO: bf16 rows (s*8+b), cols h*32+d
// ---------------------------------------------------------------------------
__global__ void __launch_bounds__(256) k_attn(
    const unsigned short* __restrict__ Qp,
    const unsigned short* __restrict__ Kp,
    const unsigned short* __restrict__ Vt,
    unsigned short* __restrict__ AO)
{
  __shared__ unsigned short Pst[8][512];           // per-wave 16x32 P staging
  const int bh = blockIdx.x;
  const int b  = bh >> 3, h = bh & 7;
  const unsigned short* Q  = Qp + (size_t)bh * S_LEN * 32;
  const unsigned short* Kb = Kp + (size_t)bh * S_LEN * 32;
  const unsigned short* Vb = Vt + (size_t)bh * 32 * S_LEN;
  const int w    = threadIdx.x >> 5;
  const int lane = threadIdx.x & 31;
  const int half = lane >> 4;
  const int r16  = lane & 15;
  unsigned short* pp = &Pst[w][0];

  for (int qt = w; qt < 64; qt += 8){
    Frag fq;
    { const unsigned short* p = Q + (size_t)(qt * 16 + r16) * 32 + half * 8;
      fq.lo = *(const u32x4*)p; fq.hi = *(const u32x4*)(p + 16); }
    v16bf vq = fragv(fq);
    float m[8], l[8];
    v8f accA{}, accB{};
    #pragma unroll
    for (int i = 0; i < 8; i++){ m[i] = -1e30f; l[i] = 0.f; }

    for (int kt = 0; kt < 32; kt++){
      const int key0 = kt * 32;
      v8f s0{}, s1{};
      { Frag fk; const unsigned short* p = Kb + (size_t)(key0 + r16) * 32 + half * 8;
        fk.lo = *(const u32x4*)p; fk.hi = *(const u32x4*)(p + 16);
        s0 = __builtin_amdgcn_wmma_f32_16x16x32_bf16(false, vq, false, fragv(fk),
                                                     (short)0, s0, false, false); }
      { Frag fk; const unsigned short* p = Kb + (size_t)(key0 + 16 + r16) * 32 + half * 8;
        fk.lo = *(const u32x4*)p; fk.hi = *(const u32x4*)(p + 16);
        s1 = __builtin_amdgcn_wmma_f32_16x16x32_bf16(false, vq, false, fragv(fk),
                                                     (short)0, s1, false, false); }
      #pragma unroll
      for (int i = 0; i < 8; i++){
        float mx = fmaxf(s0[i], s1[i]);
        mx = fmaxf(mx, __shfl_xor(mx, 1, 32));
        mx = fmaxf(mx, __shfl_xor(mx, 2, 32));
        mx = fmaxf(mx, __shfl_xor(mx, 4, 32));
        mx = fmaxf(mx, __shfl_xor(mx, 8, 32));
        const float nm = fmaxf(m[i], mx);
        const float sc = __expf(m[i] - nm);
        const float p0 = __expf(s0[i] - nm);
        const float p1 = __expf(s1[i] - nm);
        float rs = p0 + p1;
        rs += __shfl_xor(rs, 1, 32);
        rs += __shfl_xor(rs, 2, 32);
        rs += __shfl_xor(rs, 4, 32);
        rs += __shfl_xor(rs, 8, 32);
        l[i] = l[i] * sc + rs;
        m[i] = nm;
        accA[i] *= sc; accB[i] *= sc;
        const int rowp = i + half * 8;
        pp[rowp * 32 + r16]      = f2bfu(p0);
        pp[rowp * 32 + r16 + 16] = f2bfu(p1);
      }
      asm volatile("s_wait_dscnt 0" ::: "memory");     // P staged in LDS
      Frag fp;
      { const unsigned short* p = pp + r16 * 32 + half * 8;
        fp.lo = *(const u32x4*)p; fp.hi = *(const u32x4*)(p + 16); }
      v16bf vp = fragv(fp);
      { Frag fv; const unsigned short* p = Vb + (size_t)r16 * S_LEN + key0 + half * 8;
        fv.lo = *(const u32x4*)p; fv.hi = *(const u32x4*)(p + 16);
        accA = __builtin_amdgcn_wmma_f32_16x16x32_bf16(false, vp, false, fragv(fv),
                                                       (short)0, accA, false, false); }
      { Frag fv; const unsigned short* p = Vb + (size_t)(16 + r16) * S_LEN + key0 + half * 8;
        fv.lo = *(const u32x4*)p; fv.hi = *(const u32x4*)(p + 16);
        accB = __builtin_amdgcn_wmma_f32_16x16x32_bf16(false, vp, false, fragv(fv),
                                                       (short)0, accB, false, false); }
    }
    #pragma unroll
    for (int i = 0; i < 8; i++){
      const int s   = qt * 16 + i + half * 8;
      const float inv = 1.f / l[i];
      size_t base = ((size_t)s * 8 + b) * 256 + (size_t)h * 32;
      AO[base + r16]      = f2bfu(accA[i] * inv);
      AO[base + r16 + 16] = f2bfu(accB[i] * inv);
    }
  }
}

// ------------------------- LayerNorm (wave per row) -------------------------
__global__ void __launch_bounds__(128) k_ln(
    const float* __restrict__ X, const float* __restrict__ g, const float* __restrict__ b,
    int N, int R_, int preGelu, int post,  // post: 0 none, 1 sigmoid, 2 gelu
    float* __restrict__ outF, unsigned short* __restrict__ outB)
{
  const int w    = threadIdx.x >> 5;
  const int lane = threadIdx.x & 31;
  const int row  = blockIdx.x * 4 + w;
  if (row >= R_) return;
  const float* x = X + (size_t)row * N;
  float s = 0.f, s2 = 0.f;
  for (int c = lane; c < N; c += 32){
    float v = x[c];
    if (preGelu) v = geluf(v);
    s += v; s2 += v * v;
  }
  for (int d = 1; d < 32; d <<= 1){ s += __shfl_xor(s, d, 32); s2 += __shfl_xor(s2, d, 32); }
  const float mean = s / N;
  const float var  = s2 / N - mean * mean;
  const float inv  = rsqrtf(var + 1e-5f);
  for (int c = lane; c < N; c += 32){
    float v = x[c];
    if (preGelu) v = geluf(v);
    v = (v - mean) * inv * g[c] + b[c];
    if (post == 1) v = sigm(v);
    else if (post == 2) v = geluf(v);
    if (outF) outF[(size_t)row * N + c] = v;
    if (outB) outB[(size_t)row * N + c] = f2bfu(v);
  }
}

// ------------------------- small elementwise kernels ------------------------
__global__ void k_bn(const float* __restrict__ inp, const float* g, const float* b,
                     const float* m, const float* v, unsigned short* out, size_t n){
  size_t i = (size_t)blockIdx.x * 256 + threadIdx.x; if (i >= n) return;
  int c = (int)(i & 63);
  out[i] = f2bfu((inp[i] - m[c]) * rsqrtf(v[c] + 1e-5f) * g[c] + b[c]);
}
__global__ void k_gelu_bf(const float* __restrict__ in, unsigned short* out, size_t n){
  size_t i = (size_t)blockIdx.x * 256 + threadIdx.x; if (i >= n) return;
  out[i] = f2bfu(geluf(in[i]));
}
__global__ void k_gelu_x(float* __restrict__ x, unsigned short* out, size_t n){
  size_t i = (size_t)blockIdx.x * 256 + threadIdx.x; if (i >= n) return;
  float v = geluf(x[i]); x[i] = v; out[i] = f2bfu(v);
}
__global__ void k_transpose_t0(const float* __restrict__ x, unsigned short* t, size_t n){
  size_t i = (size_t)blockIdx.x * 256 + threadIdx.x; if (i >= n) return;
  int c = (int)(i & 255); size_t r = i >> 8; int b = (int)(r & 7); size_t s = r >> 3;
  t[((size_t)b * S_LEN + s) * 256 + c] = f2bfu(x[i]);
}
__global__ void k_tcn_comb(const float* __restrict__ raw, const float* __restrict__ resF,
                           const unsigned short* __restrict__ resB, unsigned short* out, size_t n){
  size_t i = (size_t)blockIdx.x * 256 + threadIdx.x; if (i >= n) return;
  float r = resF ? resF[i] : bf2f(resB[i]);
  out[i] = f2bfu(geluf(geluf(raw[i]) + r));
}
__global__ void k_pack_qkv(const float* __restrict__ qkv, unsigned short* Qp,
                           unsigned short* Kp, unsigned short* Vt, size_t n){
  size_t i = (size_t)blockIdx.x * 256 + threadIdx.x; if (i >= n) return;
  int d = (int)(i & 31); size_t t = i >> 5; int s = (int)(t & 1023); int bh = (int)(t >> 10);
  int b = bh >> 3, h = bh & 7;
  size_t base = ((size_t)s * 8 + b) * 768 + (size_t)h * 32 + d;
  Qp[i] = f2bfu(qkv[base] * 0.17677669529663689f);   // 1/sqrt(32)
  Kp[i] = f2bfu(qkv[base + 256]);
  Vt[((size_t)bh * 32 + d) * S_LEN + s] = f2bfu(qkv[base + 512]);
}
__global__ void k_pack_cg(const unsigned short* __restrict__ t4, const float* __restrict__ mo,
                          float* cg, unsigned short* cgbf, size_t n){
  size_t i = (size_t)blockIdx.x * 256 + threadIdx.x; if (i >= n) return;
  int c = (int)(i & 511); size_t r = i >> 9; int b = (int)(r & 7); size_t s = r >> 3;
  float v = (c < 256) ? bf2f(t4[((size_t)b * S_LEN + s) * 256 + c])
                      : mo[r * 256 + (c - 256)];
  cg[i] = v; cgbf[i] = f2bfu(v);
}
__global__ void k_comb(const float* __restrict__ gv, const float* __restrict__ cg,
                       const float* __restrict__ at, unsigned short* comb, size_t n){
  size_t i = (size_t)blockIdx.x * 256 + threadIdx.x; if (i >= n) return;
  int c = (int)(i % 768); size_t r = i / 768;
  float v = (c < 512) ? gv[r * 512 + c] * cg[r * 512 + c] : at[r * 256 + (c - 512)];
  comb[i] = f2bfu(v);
}
__global__ void k_add_bf(const float* __restrict__ a, const float* __restrict__ b,
                         unsigned short* out, size_t n){
  size_t i = (size_t)blockIdx.x * 256 + threadIdx.x; if (i >= n) return;
  out[i] = f2bfu(a[i] + b[i]);
}
__global__ void k_head(const unsigned short* __restrict__ o1b, const float* __restrict__ w2,
                       const float* __restrict__ b2, float* y2, size_t n){
  size_t i = (size_t)blockIdx.x * 256 + threadIdx.x; if (i >= n) return;
  int o = (int)(i & 1); size_t r = i >> 1;
  float acc = b2[o];
  const unsigned short* row = o1b + r * 128;
  for (int k2 = 0; k2 < 128; k2++) acc += bf2f(row[k2]) * w2[k2 * 2 + o];
  y2[i] = tanhf(acc);
}
__global__ void k_smooth(const float* __restrict__ y2, float* out, size_t n){
  size_t i = (size_t)blockIdx.x * 256 + threadIdx.x; if (i >= n) return;
  int o = (int)(i & 1); size_t t = i >> 1; int b = (int)(t & 7); long s = (long)(t >> 3);
  float acc = 0.f;
  for (int ds = -2; ds <= 2; ds++){
    long ss = s + ds;
    if (ss >= 0 && ss < S_LEN) acc += y2[(((size_t)ss * 8 + b) << 1) + o];
  }
  out[i] = acc * 0.2f;
}

// ------------------------- weight prep kernels ------------------------------
__global__ void k_wT(const float* __restrict__ W, unsigned short* WT, int K, int N){
  size_t i = (size_t)blockIdx.x * 256 + threadIdx.x; if (i >= (size_t)K * N) return;
  int n = (int)(i / K), k = (int)(i % K);
  WT[(size_t)n * K + k] = f2bfu(W[(size_t)k * N + n]);
}
__global__ void k_wTap(const float* __restrict__ W, unsigned short* dst,
                       int Cout, int Cin, int KS, int tap){
  size_t i = (size_t)blockIdx.x * 256 + threadIdx.x; if (i >= (size_t)Cout * Cin) return;
  dst[i] = f2bfu(W[i * KS + tap]);
}
__global__ void k_cvt(const float* __restrict__ src, unsigned short* dst, size_t n){
  size_t i = (size_t)blockIdx.x * 256 + threadIdx.x; if (i >= n) return;
  dst[i] = f2bfu(src[i]);
}
__global__ void k_aneg(const float* __restrict__ alog, float* A, size_t n){
  size_t i = (size_t)blockIdx.x * 256 + threadIdx.x; if (i >= n) return;
  A[i] = -expf(alog[i]);
}

// ------------------------- mamba VALU kernels -------------------------------
__global__ void k_mamba_conv(const float* __restrict__ xz, const float* __restrict__ cw,
                             const float* __restrict__ cb, float* xc, unsigned short* xcb, size_t n){
  size_t i = (size_t)blockIdx.x * 256 + threadIdx.x; if (i >= n) return;
  int d = (int)(i & 127); size_t rr = i >> 7; int l = (int)(rr & 7); size_t bm = rr >> 3;
  float acc = cb[d];
  #pragma unroll
  for (int k = 0; k < 4; k++){
    int ls = l - 3 + k;
    if (ls >= 0) acc += cw[d * 4 + k] * xz[((bm << 3) + ls) * 256 + d];
  }
  float s = acc * sigm(acc);        // silu
  xc[i] = s; xcb[i] = f2bfu(s);
}
__global__ void k_mamba_dt(const float* __restrict__ xp, const float* __restrict__ dtw,
                           const float* __restrict__ dtb, float* dt, size_t n){
  size_t i = (size_t)blockIdx.x * 256 + threadIdx.x; if (i >= n) return;
  int d = (int)(i & 127); size_t r = i >> 7;
  float acc = dtb[d];
  #pragma unroll
  for (int j = 0; j < 4; j++) acc += xp[r * 260 + j] * dtw[j * 128 + d];
  dt[i] = (acc > 20.f) ? acc : log1pf(expf(acc));    // softplus
}
__global__ void __launch_bounds__(128) k_mamba_scan(
    const float* __restrict__ xz, const float* __restrict__ xc,
    const float* __restrict__ dtb, const float* __restrict__ xp,
    const float* __restrict__ Aneg, const float* __restrict__ D,
    unsigned short* __restrict__ ymid)
{
  __shared__ float Bs[8][128], Cs[8][128];
  const int bm = blockIdx.x;
  const int d  = threadIdx.x;
  for (int l = 0; l < 8; l++){
    Bs[l][d] = xp[((size_t)bm * 8 + l) * 260 + 4 + d];
    Cs[l][d] = xp[((size_t)bm * 8 + l) * 260 + 132 + d];
  }
  __syncthreads();
  float dtv[8], uv[8], y[8];
  #pragma unroll
  for (int l = 0; l < 8; l++){
    size_t rr = (size_t)bm * 8 + l;
    dtv[l] = dtb[rr * 128 + d];
    uv[l]  = xc[rr * 128 + d];
    y[l] = 0.f;
  }
  for (int n = 0; n < 128; n++){
    float a = Aneg[(size_t)d * 128 + n];
    float h = 0.f;
    #pragma unroll
    for (int l = 0; l < 8; l++){
      h = __expf(dtv[l] * a) * h + dtv[l] * uv[l] * Bs[l][n];
      y[l] += h * Cs[l][n];
    }
  }
  const float Dd = D[d];
  #pragma unroll
  for (int l = 0; l < 8; l++){
    size_t rr = (size_t)bm * 8 + l;
    float z = xz[rr * 256 + 128 + d];
    ymid[rr * 128 + d] = f2bfu((y[l] + uv[l] * Dd) * (z * sigm(z)));
  }
}

// --------------------------------- host -------------------------------------
static inline unsigned gblk(size_t n){ return (unsigned)((n + 255) / 256); }

static void gemm(hipStream_t st, const unsigned short* A, int lda, const unsigned short* WT,
                 const float* bias, float* C, int ldc, int M, int N, int K,
                 int rowShift = 0, int seg = (1 << 30), int accum = 0)
{
  if (rowShift == 0 && (M % 256) == 0 && (N % 64) == 0){
    dim3 grid((unsigned)(N / 64), (unsigned)(M / 256));
    k_gemm_fast<<<grid, 256, 0, st>>>(A, lda, WT, bias, C, ldc, K, accum);
  } else {
    dim3 grid((unsigned)((N + 63) / 64), (unsigned)((M + 127) / 128));
    k_gemm_g<<<grid, 256, 0, st>>>(A, lda, WT, bias, C, ldc, M, N, K, rowShift, seg, accum);
  }
}

extern "C" void kernel_launch(void* const* d_in, const int* in_sizes, int n_in,
                              void* d_out, int out_size, void* d_ws, size_t ws_size,
                              hipStream_t stream)
{
  (void)in_sizes; (void)out_size;
  if (n_in < 91) return;
  const size_t MB = 1u << 20;
  const size_t R  = RROWS;

  const float* inp = (const float*)d_in[0];
  auto pr = [&](int i){ return (const float*)d_in[i]; };

  // ---- workspace layout ----
  unsigned char* ws = (unsigned char*)d_ws;
  size_t off = 0;
  auto alloc = [&](size_t bytes)->void*{
    off = (off + 255) & ~(size_t)255;
    void* p = ws + off; off += bytes; return p;
  };
  unsigned short* wp   = (unsigned short*)alloc(24 * MB);   // bf16 weight arena
  float*          x    = (float*)         alloc(R * 256 * 4);
  unsigned short* xbf  = (unsigned short*)alloc(R * 256 * 2);
  unsigned short* t4   = (unsigned short*)alloc(R * 256 * 2);
  float*          mamo = (float*)         alloc(R * 256 * 4);
  float*          atto = (float*)         alloc(R * 256 * 4);
  float*          cg   = (float*)         alloc(R * 512 * 4);
  unsigned short* cgbf = (unsigned short*)alloc(R * 512 * 2);
  unsigned char*  AR   = (unsigned char*) alloc(60 * MB);   // phase-reused arena
  if (off > ws_size) return;
  auto arf = [&](size_t mb){ return (float*)         (AR + mb * MB); };
  auto arb = [&](size_t mb){ return (unsigned short*)(AR + mb * MB); };

  // ---- weight prep (fp32 -> bf16, [N][K] transposed) ----
  size_t woff = 0;
  auto wt = [&](const float* W, int K, int N)->unsigned short*{
    unsigned short* d = wp + woff; woff += (size_t)K * N;
    k_wT<<<gblk((size_t)K * N), 256, 0, stream>>>(W, d, K, N);
    return d;
  };
  auto wtap = [&](const float* W, int Cout, int Cin, int tap)->unsigned short*{
    unsigned short* d = wp + woff; woff += (size_t)Cout * Cin;
    k_wTap<<<gblk((size_t)Cout * Cin), 256, 0, stream>>>(W, d, Cout, Cin, 5, tap);
    return d;
  };
  auto wcv = [&](const float* W, size_t n)->unsigned short*{
    unsigned short* d = wp + woff; woff += n;
    k_cvt<<<gblk(n), 256, 0, stream>>>(W, d, n);
    return d;
  };

  unsigned short* w0T = wt(pr(5),  64, 128);
  unsigned short* w1T = wt(pr(9), 128, 256);
  const int cins[4]  = {256, 512, 512, 512};
  const int couts[4] = {512, 512, 512, 256};
  const int dils[4]  = {1, 2, 4, 8};
  const int tb[4]    = {11, 17, 21, 25};
  unsigned short* W1t[4][5]; unsigned short* W2t[4][5]; unsigned short* dwT[4] = {0,0,0,0};
  for (int blk = 0; blk < 4; blk++){
    for (int kk = 0; kk < 5; kk++) W1t[blk][kk] = wtap(pr(tb[blk]),     couts[blk], cins[blk],  kk);
    for (int kk = 0; kk < 5; kk++) W2t[blk][kk] = wtap(pr(tb[blk] + 2), couts[blk], couts[blk], kk);
  }
  dwT[0] = wcv(pr(15), (size_t)512 * 256);
  dwT[3] = wcv(pr(29), (size_t)256 * 512);
  unsigned short* inWT[4]; unsigned short* xpT[4]; unsigned short* outWT[4];
  for (int i = 0; i < 4; i++){
    inWT[i]  = wt(pr(31 + 9 * i),      64, 256);
    xpT[i]   = wt(pr(31 + 9 * i + 3), 128, 260);
    outWT[i] = wt(pr(31 + 9 * i + 8), 128, 64);
  }
  unsigned short* ainT  = wt(pr(67), 256, 768);
  unsigned short* aoutT = wt(pr(69), 256, 256);
  unsigned short* gateT = wt(pr(71), 512, 512);
  unsigned short* fusT  = wt(pr(75), 768, 512);
  unsigned short* resT  = wt(pr(79), 256, 512);
  unsigned short* o0T   = wt(pr(81), 512, 256);
  unsigned short* o1T   = wt(pr(85), 256, 128);

  // ---- phase 0: input MLP ----
  unsigned short* a0 = arb(0);
  float*          g0 = arf(2);
  unsigned short* a1 = arb(6);
  k_bn<<<gblk(R * 64), 256, 0, stream>>>(inp, pr(1), pr(2), pr(3), pr(4), a0, R * 64);
  gemm(stream, a0, 64, w0T, pr(6), g0, 128, (int)R, 128, 64);
  k_ln<<<RROWS / 4, 128, 0, stream>>>(g0, pr(7), pr(8), 128, RROWS, 1, 0, nullptr, a1);
  gemm(stream, a1, 128, w1T, pr(10), x, 256, (int)R, 256, 128);
  k_gelu_x<<<gblk(R * 256), 256, 0, stream>>>(x, xbf, R * 256);

  // ---- phase 1: TCN (dilated causal convs as 5 shifted GEMM taps) ----
  unsigned short* tA   = arb(0);
  unsigned short* tB   = arb(8);
  float*          tmp  = arf(16);
  unsigned short* gbf  = arb(32);
  float*          resb = arf(40);
  k_transpose_t0<<<gblk(R * 256), 256, 0, stream>>>(x, tA, R * 256);
  unsigned short* tcur = tA;
  for (int blk = 0; blk < 4; blk++){
    const int ci = cins[blk], co = couts[blk], dil = dils[blk], pb = tb[blk];
    for (int kk = 0; kk < 5; kk++)
      gemm(stream, tcur, ci, W1t[blk][kk], kk == 0 ? pr(pb + 1) : nullptr,
           tmp, co, (int)R, co, ci, (4 - kk) * dil, S_LEN, kk > 0);
    k_gelu_bf<<<gblk(R * (size_t)co), 256, 0, stream>>>(tmp, gbf, R * (size_t)co);
    for (int kk = 0; kk < 5; kk++)
      gemm(stream, gbf, co, W2t[blk][kk], kk == 0 ? pr(pb + 3) : nullptr,
           tmp, co, (int)R, co, co, (4 - kk) * dil, S_LEN, kk > 0);
    const float* resF = nullptr; const unsigned short* resB = nullptr;
    if (blk == 0 || blk == 3){
      gemm(stream, tcur, ci, dwT[blk], pr(blk == 0 ? 16 : 30), resb, co, (int)R, co, ci);
      resF = resb;
    } else resB = tcur;
    unsigned short* tnext = (tcur == tA) ? tB : tA;
    k_tcn_comb<<<gblk(R * (size_t)co), 256, 0, stream>>>(tmp, resF, resB, tnext, R * (size_t)co);
    tcur = tnext;
  }
  hipMemcpyAsync(t4, tcur, R * 256 * 2, hipMemcpyDeviceToDevice, stream);

  // ---- phase 2: 4 Mamba modules (scan length = 8) ----
  float*          xz   = arf(0);
  float*          xcf  = arf(8);
  unsigned short* xcb  = arb(12);
  float*          xp   = arf(14);
  float*          dtv  = arf(23);
  unsigned short* ymid = arb(27);
  float*          Aneg = arf(29);
  for (int i = 0; i < 4; i++){
    const int base = 31 + 9 * i;
    gemm(stream, xbf + i * 64, 256, inWT[i], nullptr, xz, 256, (int)R, 256, 64);
    k_mamba_conv<<<gblk(R * 128), 256, 0, stream>>>(xz, pr(base + 1), pr(base + 2), xcf, xcb, R * 128);
    gemm(stream, xcb, 128, xpT[i], nullptr, xp, 260, (int)R, 260, 128);
    k_mamba_dt<<<gblk(R * 128), 256, 0, stream>>>(xp, pr(base + 4), pr(base + 5), dtv, R * 128);
    k_aneg<<<gblk((size_t)128 * 128), 256, 0, stream>>>(pr(base + 6), Aneg, (size_t)128 * 128);
    k_mamba_scan<<<S_LEN, 128, 0, stream>>>(xz, xcf, dtv, xp, Aneg, pr(base + 7), ymid);
    gemm(stream, ymid, 128, outWT[i], nullptr, mamo + i * 64, 256, (int)R, 64, 128);
  }

  // ---- phase 3: attention ----
  float*          qkv  = arf(0);
  unsigned short* Qp   = arb(24);
  unsigned short* Kpb  = arb(28);
  unsigned short* Vtb  = arb(32);
  unsigned short* aobf = arb(36);
  gemm(stream, xbf, 256, ainT, pr(68), qkv, 768, (int)R, 768, 256);
  k_pack_qkv<<<gblk((size_t)64 * S_LEN * 32), 256, 0, stream>>>(qkv, Qp, Kpb, Vtb,
                                                                (size_t)64 * S_LEN * 32);
  k_attn<<<64, 256, 0, stream>>>(Qp, Kpb, Vtb, aobf);
  gemm(stream, aobf, 256, aoutT, pr(70), atto, 256, (int)R, 256, 256);

  // ---- phase 4: gate / fusion / residual / output MLP / smoothing ----
  float*          glin  = arf(0);
  float*          gatev = arf(16);
  unsigned short* comb  = arb(32);
  float*          flin  = arf(44);
  k_pack_cg<<<gblk(R * 512), 256, 0, stream>>>(t4, mamo, cg, cgbf, R * 512);
  gemm(stream, cgbf, 512, gateT, pr(72), glin, 512, (int)R, 512, 512);
  k_ln<<<RROWS / 4, 128, 0, stream>>>(glin, pr(73), pr(74), 512, RROWS, 0, 1, gatev, nullptr);
  k_comb<<<gblk(R * 768), 256, 0, stream>>>(gatev, cg, atto, comb, R * 768);
  gemm(stream, comb, 768, fusT, pr(76), flin, 512, (int)R, 512, 768);
  float*          fused = arf(0);
  float*          rlin  = arf(16);
  unsigned short* hbf   = arb(44);
  k_ln<<<RROWS / 4, 128, 0, stream>>>(flin, pr(77), pr(78), 512, RROWS, 0, 2, fused, nullptr);
  gemm(stream, xbf, 256, resT, pr(80), rlin, 512, (int)R, 512, 256);
  k_add_bf<<<gblk(R * 512), 256, 0, stream>>>(fused, rlin, hbf, R * 512);
  float*          o0   = arf(16);
  unsigned short* o0b  = arb(24);
  float*          o1   = arf(28);
  unsigned short* o1b  = arb(32);
  float*          y2   = arf(34);
  gemm(stream, hbf, 512, o0T, pr(82), o0, 256, (int)R, 256, 512);
  k_ln<<<RROWS / 4, 128, 0, stream>>>(o0, pr(83), pr(84), 256, RROWS, 1, 0, nullptr, o0b);
  gemm(stream, o0b, 256, o1T, pr(86), o1, 128, (int)R, 128, 256);
  k_ln<<<RROWS / 4, 128, 0, stream>>>(o1, pr(87), pr(88), 128, RROWS, 1, 0, nullptr, o1b);
  k_head<<<gblk(R * 2), 256, 0, stream>>>(o1b, pr(89), pr(90), y2, R * 2);
  k_smooth<<<gblk((size_t)S_LEN * BATCH * 2), 256, 0, stream>>>(y2, (float*)d_out,
                                                                (size_t)S_LEN * BATCH * 2);
}
